// MovingAverageGatedAttention_38010460570385
// MI455X (gfx1250) — matrix operand order repaired and checked by reading, not compiled, not run on hardware
//
#include <hip/hip_runtime.h>
#include <hip/hip_bf16.h>
#include <math.h>

typedef __bf16 bf16;
typedef __attribute__((ext_vector_type(16))) __bf16 v16bf;
typedef __attribute__((ext_vector_type(8)))  float   v8f;
typedef __attribute__((ext_vector_type(4)))  int     v4i_;

#define L_ 2048
#define B_ 8
#define D_ 1024
#define Z_ 128
#define H_ 2048
#define N_ 16
#define NOUT_ 4224   /* Z + H + 2D */
#define MAXPOS_ 2048

#if defined(__gfx1250__) && __has_builtin(__builtin_amdgcn_global_load_async_to_lds_b128) && __has_builtin(__builtin_amdgcn_s_wait_asynccnt)
#define HAVE_ASYNC_LDS 1
typedef __attribute__((address_space(1))) v4i_* g4p;
typedef __attribute__((address_space(3))) v4i_* l4p;
#else
#define HAVE_ASYNC_LDS 0
#endif

__device__ __forceinline__ float sigmoidf_(float x){ return 1.0f/(1.0f+__expf(-x)); }
__device__ __forceinline__ float siluf_(float x){ return x*sigmoidf_(x); }

// ---------------- f32 -> bf16 transpose-convert (weights -> W^T, n-major) ----------------
__global__ __launch_bounds__(256) void cvt_t_kernel(const float* __restrict__ in,
                                                    bf16* __restrict__ out,
                                                    int K, int N){
  __shared__ float tile[32][33];
  const int k0 = blockIdx.y*32, n0 = blockIdx.x*32;
  const int tx = threadIdx.x & 31, ty = threadIdx.x >> 5;   // 32 x 8
  #pragma unroll
  for (int i=0;i<32;i+=8){
    int k = k0+ty+i, n = n0+tx;
    tile[ty+i][tx] = (k<K && n<N) ? in[(size_t)k*N + n] : 0.f;
  }
  __syncthreads();
  #pragma unroll
  for (int i=0;i<32;i+=8){
    int n = n0+ty+i, k = k0+tx;
    if (n<N && k<K) out[(size_t)n*K + k] = (bf16)tile[tx][ty+i];
  }
}

// ---------------- LayerNorm: one 256-thread block per row (D=1024) ----------------
__global__ __launch_bounds__(256) void ln_kernel(const float* __restrict__ x,
                                                 const float* __restrict__ w,
                                                 const float* __restrict__ b,
                                                 bf16* __restrict__ xn){
  __shared__ float red[256];
  const int row = blockIdx.x;
  const float* xr = x + (size_t)row*D_;
  float v4[4]; float s = 0.f;
  #pragma unroll
  for (int i=0;i<4;++i){ v4[i] = xr[threadIdx.x + i*256]; s += v4[i]; }
  red[threadIdx.x] = s; __syncthreads();
  for (int o=128;o>0;o>>=1){ if (threadIdx.x<o) red[threadIdx.x]+=red[threadIdx.x+o]; __syncthreads(); }
  const float mu = red[0]*(1.0f/D_); __syncthreads();
  s = 0.f;
  #pragma unroll
  for (int i=0;i<4;++i){ float d=v4[i]-mu; s += d*d; }
  red[threadIdx.x] = s; __syncthreads();
  for (int o=128;o>0;o>>=1){ if (threadIdx.x<o) red[threadIdx.x]+=red[threadIdx.x+o]; __syncthreads(); }
  const float rstd = rsqrtf(red[0]*(1.0f/D_) + 1e-5f);
  #pragma unroll
  for (int i=0;i<4;++i){
    int c = threadIdx.x + i*256;
    xn[(size_t)row*D_ + c] = (bf16)((v4[i]-mu)*rstd*w[c] + b[c]);
  }
}

// ---------------- EMA conv as 16 first-order recurrences per (b,d) ----------------
__global__ __launch_bounds__(256) void ema_kernel(const bf16* __restrict__ xn,
                                                  const float* __restrict__ delta,
                                                  const float* __restrict__ alpha,
                                                  const float* __restrict__ beta,
                                                  const float* __restrict__ gamma,
                                                  const float* __restrict__ omega,
                                                  bf16* __restrict__ mx){
  int idx = blockIdx.x*blockDim.x + threadIdx.x;
  if (idx >= B_*D_) return;
  const int b = idx / D_;
  const int d = idx % D_;
  float qv[N_], wv[N_], gv[N_], sv[N_];
  const float scale = 0.25f; // 1/sqrt(N)
  #pragma unroll
  for (int n=0;n<N_;++n){
    float p = sigmoidf_(delta[d*N_+n]);
    qv[n] = 1.0f - p*sigmoidf_(alpha[d*N_+n]);
    wv[n] = p*beta[d*N_+n];
    gv[n] = gamma[d*N_+n]*scale;
    sv[n] = 0.f;
  }
  const float om = omega[d];
  for (int l=0;l<L_;++l){
    size_t off = ((size_t)l*B_ + b)*D_ + d;
    float xv = (float)xn[off];
    float acc = 0.f;
    #pragma unroll
    for (int n=0;n<N_;++n){ sv[n] = qv[n]*sv[n] + wv[n]*xv; acc += gv[n]*sv[n]; }
    mx[off] = (bf16)siluf_(acc + om*xv);
  }
}

// ---------------- generic tiled WMMA bf16 GEMM with fused epilogues ----------------
// C = A(M x K, row-major) * B^T(N x K, n-major) ; K-tile = 64, block tile 128x128, 8 wave32 waves.
enum { M_V = 0, M_BASE = 1, M_QK = 2, M_AV = 3, M_FINAL = 4 };

struct GemmP {
  const bf16* A; const bf16* A2; const bf16* Bt;   // Bt is N x K (n-major)
  long long lda, ldb, batchA, batchB;
  const float* bias;
  float* out_f; bf16* out_bf;
  const float* relpos; const float* qkg; const float* qkb;
  bf16* qout; bf16* kout; bf16* rout; float* uout; float* hxout;
  const float* hx; const float* u; const float* xres;
  int K;
};

#define LDSS 72   /* 64 + 8 pad, bf16 elements */

// stage 64 contiguous bytes global -> LDS (async when available)
__device__ __forceinline__ void stage64(const bf16* g, bf16* l){
#if HAVE_ASYNC_LDS
  g4p ga = (g4p)g;
  l4p la = (l4p)l;
  __builtin_amdgcn_global_load_async_to_lds_b128(ga, la, 0, 0);
  __builtin_amdgcn_global_load_async_to_lds_b128(ga, la, 16, 0);
  __builtin_amdgcn_global_load_async_to_lds_b128(ga, la, 32, 0);
  __builtin_amdgcn_global_load_async_to_lds_b128(ga, la, 48, 0);
#else
  #pragma unroll
  for (int j=0;j<32;++j) l[j] = g[j];
#endif
}

template<int MODE, bool MULA2>
__global__ __launch_bounds__(256) void gemm_kernel(GemmP p){
  __shared__ bf16 As[128*LDSS];   // 128 rows x 64 k
  __shared__ bf16 Bs[128*LDSS];   // 128 n-rows x 64 k
  const int tid   = threadIdx.x;
  const int lane  = tid & 31;
  const int wave  = tid >> 5;     // 8 wave32 waves
  const int waveM = wave >> 2;    // 0..1 -> 64 rows each
  const int waveN = wave & 3;     // 0..3 -> 32 cols each
  const int lm    = lane & 15;
  const int hi    = lane >> 4;    // lane half
  const int gm0   = blockIdx.y * 128;
  const int gn0   = blockIdx.x * 128;
  const int bz    = blockIdx.z;
  const bf16* Ab = p.A  + (size_t)bz * p.batchA;
  const bf16* Bb = p.Bt + (size_t)bz * p.batchB;

  v8f acc[4][2];
  #pragma unroll
  for (int mi=0;mi<4;++mi)
    #pragma unroll
    for (int ni=0;ni<2;++ni)
      #pragma unroll
      for (int e=0;e<8;++e) acc[mi][ni][e] = 0.f;

  // staging: each thread covers 32 contiguous bf16 (half a 64-wide row)
  const int sr = tid >> 1;          // row 0..127
  const int sc = (tid & 1) * 32;    // col 0 / 32

  for (int kb = 0; kb < p.K; kb += 64){
    __syncthreads();
    {
      const bf16* srcA = Ab + (size_t)(gm0 + sr)*p.lda + kb + sc;
      if (MULA2){
        const bf16* srcA2 = p.A2 + (size_t)bz*p.batchA + (size_t)(gm0 + sr)*p.lda + kb + sc;
        #pragma unroll
        for (int j=0;j<32;++j) As[sr*LDSS + sc + j] = (bf16)((float)srcA[j]*(float)srcA2[j]);
      } else {
        stage64(srcA, &As[sr*LDSS + sc]);
      }
      const bf16* srcB = Bb + (size_t)(gn0 + sr)*p.ldb + kb + sc;
      stage64(srcB, &Bs[sr*LDSS + sc]);
    }
#if !HAVE_ASYNC_LDS
    if (kb + 64 < p.K){
      __builtin_prefetch(Ab + (size_t)(gm0 + sr)*p.lda + kb + 64 + sc, 0, 1);
      __builtin_prefetch(Bb + (size_t)(gn0 + sr)*p.ldb + kb + 64 + sc, 0, 1);
    }
#endif
#if HAVE_ASYNC_LDS
    __builtin_amdgcn_s_wait_asynccnt(0);
#endif
    __syncthreads();

    // two 32-wide k-subtiles per staged tile
    #pragma unroll
    for (int ks=0; ks<64; ks+=32){
      v16bf afr[4], bfr[2];
      #pragma unroll
      for (int mi=0;mi<4;++mi){
        const int m0 = waveM*64 + mi*16 + lm;
        #pragma unroll
        for (int i=0;i<16;++i){
          // A 16x32 frag: k = (i&7) + 8*laneHalf + 16*(i>>3) -> two contiguous 16B chunks
          const int k = ks + (i & 7) + (hi << 3) + ((i >> 3) << 4);
          afr[mi][i] = As[m0*LDSS + k];
        }
      }
      #pragma unroll
      for (int ni=0;ni<2;++ni){
        const int n0 = waveN*32 + ni*16 + lm;
        #pragma unroll
        for (int i=0;i<16;++i){
          // B frag from n-major tile: k = (i&7) + 8*laneHalf + 16*(i>>3) -> two contiguous 16B chunks
          const int k = ks + (i & 7) + (hi << 3) + ((i >> 3) << 4);
          bfr[ni][i] = Bs[n0*LDSS + k];
        }
      }
      #pragma unroll
      for (int mi=0;mi<4;++mi)
        #pragma unroll
        for (int ni=0;ni<2;++ni)
          acc[mi][ni] = __builtin_amdgcn_wmma_f32_16x16x32_bf16(
              false, afr[mi], false, bfr[ni], (short)0, acc[mi][ni], false, false);
    }
  }

  // ---- fused epilogues. C layout: lane -> n = lane&15, VGPR v -> m = v + 8*(lane>>4) ----
  #pragma unroll
  for (int mi=0;mi<4;++mi)
    #pragma unroll
    for (int ni=0;ni<2;++ni){
      const int gn = gn0 + waveN*32 + ni*16 + lm;
      #pragma unroll
      for (int v=0; v<8; ++v){
        const int gm = gm0 + waveM*64 + mi*16 + v + (hi << 3);
        float val = acc[mi][ni][v];
        if (MODE == M_V){
          // v = silu(xn@v_w + v_b), stored (b, h, l) so attn@v streams it n-major
          float s = siluf_(val + p.bias[gn]);
          int l = gm >> 3, b = gm & 7;
          p.out_bf[((size_t)b*H_ + gn)*L_ + l] = (bf16)s;
        } else if (MODE == M_BASE){
          float t = val + p.bias[gn];
          int l = gm >> 3, b = gm & 7;
          if (gn < D_){
            p.uout[(size_t)gm*D_ + gn] = sigmoidf_(t);                    // u = sigmoid
          } else if (gn < D_ + Z_){
            float z = siluf_(t); int zi = gn - D_;
            p.qout[((size_t)b*L_ + l)*Z_ + zi] = (bf16)(z*p.qkg[zi]      + p.qkb[zi]);
            p.kout[((size_t)b*L_ + l)*Z_ + zi] = (bf16)(z*p.qkg[Z_ + zi] + p.qkb[Z_ + zi]);
          } else if (gn < D_ + Z_ + H_){
            p.rout[(size_t)gm*H_ + (gn - D_ - Z_)] = (bf16)siluf_(t);     // r gate
          } else {
            p.hxout[(size_t)gm*D_ + (gn - D_ - Z_ - H_)] = t;             // hx raw
          }
        } else if (MODE == M_QK){
          // attn = laplace(q.k^T/L + rel_pos[MAXPOS-1 + j - i])
          float t = val*(1.0f/L_) + p.relpos[MAXPOS_ - 1 + gn - gm];
          float a = 0.5f*(1.0f + erff((t - 0.70710678f)*2.5066282746f));
          p.out_bf[(size_t)bz*L_*L_ + (size_t)gm*L_ + gn] = (bf16)a;
        } else if (MODE == M_AV){
          // store h in (l*B+b) row-major so the final GEMM streams it directly
          p.out_bf[((size_t)gm*B_ + bz)*H_ + gn] = (bf16)val;
        } else { // M_FINAL : out = res + u*(silu(hx + (h*r)@h_w + h_b) - res)
          size_t o = (size_t)gm*D_ + gn;
          float h2  = siluf_(p.hx[o] + val + p.bias[gn]);
          float res = p.xres[o];
          p.out_f[o] = res + p.u[o]*(h2 - res);
        }
      }
    }
}

// ---------------- host launch ----------------
extern "C" void kernel_launch(void* const* d_in, const int* in_sizes, int n_in,
                              void* d_out, int out_size, void* d_ws, size_t ws_size,
                              hipStream_t stream){
  (void)in_sizes; (void)n_in; (void)out_size; (void)ws_size;
  const float* x      = (const float*)d_in[0];
  const float* delta  = (const float*)d_in[1];
  const float* alpha  = (const float*)d_in[2];
  const float* beta   = (const float*)d_in[3];
  const float* gamma  = (const float*)d_in[4];
  const float* omega  = (const float*)d_in[5];
  const float* v_w    = (const float*)d_in[6];
  const float* v_b    = (const float*)d_in[7];
  const float* mx_w   = (const float*)d_in[8];
  const float* mx_b   = (const float*)d_in[9];
  const float* h_w    = (const float*)d_in[10];
  const float* h_b    = (const float*)d_in[11];
  const float* qkg    = (const float*)d_in[12];
  const float* qkb    = (const float*)d_in[13];
  const float* relpos = (const float*)d_in[14];
  const float* ln_w   = (const float*)d_in[15];
  const float* ln_b   = (const float*)d_in[16];
  float* out = (float*)d_out;

  char* ws = (char*)d_ws; size_t off = 0;
  auto take = [&](size_t bytes)->char*{ char* p = ws + off; off += (bytes + 255) & ~(size_t)255; return p; };
  const size_t LB = (size_t)L_*B_;
  bf16*  xn   = (bf16*) take(LB*D_*2);
  bf16*  mx   = (bf16*) take(LB*D_*2);
  bf16*  vwt  = (bf16*) take((size_t)H_*D_*2);      // v_w^T  (H x D)
  bf16*  mxwt = (bf16*) take((size_t)NOUT_*D_*2);   // mx_w^T (4224 x 1024)
  bf16*  hwt  = (bf16*) take((size_t)D_*H_*2);      // h_w^T  (D x H)
  bf16*  vbt  = (bf16*) take((size_t)B_*H_*L_*2);   // v, stored (b, h, l)
  float* ubuf = (float*)take(LB*D_*4);
  bf16*  qb   = (bf16*) take((size_t)B_*L_*Z_*2);
  bf16*  kbuf = (bf16*) take((size_t)B_*L_*Z_*2);
  bf16*  rb   = (bf16*) take(LB*H_*2);
  float* hxb  = (float*)take(LB*D_*4);
  bf16*  attn = (bf16*) take((size_t)B_*L_*L_*2);
  bf16*  hbuf = (bf16*) take(LB*H_*2);

  // weights -> bf16, transposed to n-major
  cvt_t_kernel<<<dim3(H_/32,    D_/32), 256, 0, stream>>>(v_w,  vwt,  D_, H_);
  cvt_t_kernel<<<dim3(NOUT_/32, D_/32), 256, 0, stream>>>(mx_w, mxwt, D_, NOUT_);
  cvt_t_kernel<<<dim3(D_/32,    H_/32), 256, 0, stream>>>(h_w,  hwt,  H_, D_);

  // layernorm + EMA (recurrence form)
  ln_kernel <<<L_*B_, 256, 0, stream>>>(x, ln_w, ln_b, xn);
  ema_kernel<<<(B_*D_ + 255)/256, 256, 0, stream>>>(xn, delta, alpha, beta, gamma, omega, mx);

  // v = silu(xn @ v_w + v_b) -> (b,h,l) bf16
  GemmP pv = {}; pv.A = xn; pv.Bt = vwt; pv.lda = D_; pv.ldb = D_;
  pv.bias = v_b; pv.out_bf = vbt; pv.K = D_;
  gemm_kernel<M_V,false><<<dim3(H_/128, (int)(LB/128), 1), 256, 0, stream>>>(pv);

  // base = mx @ mx_w + mx_b, 4-way split epilogue
  GemmP pb = {}; pb.A = mx; pb.Bt = mxwt; pb.lda = D_; pb.ldb = D_; pb.bias = mx_b;
  pb.uout = ubuf; pb.qout = qb; pb.kout = kbuf; pb.rout = rb; pb.hxout = hxb;
  pb.qkg = qkg; pb.qkb = qkb; pb.K = D_;
  gemm_kernel<M_BASE,false><<<dim3(NOUT_/128, (int)(LB/128), 1), 256, 0, stream>>>(pb);

  // attn = laplace(q@k^T/L + bias), batched over B ; k is already n-major (l', z)
  GemmP pq = {}; pq.A = qb; pq.Bt = kbuf; pq.lda = Z_; pq.ldb = Z_;
  pq.batchA = (long long)L_*Z_; pq.batchB = (long long)L_*Z_;
  pq.relpos = relpos; pq.out_bf = attn; pq.K = Z_;
  gemm_kernel<M_QK,false><<<dim3(L_/128, L_/128, B_), 256, 0, stream>>>(pq);

  // h = attn @ v ; v stored (b,h,l) = n-major, out stored (l*B+b, H)
  GemmP pa = {}; pa.A = attn; pa.Bt = vbt; pa.lda = L_; pa.ldb = L_;
  pa.batchA = (long long)L_*L_; pa.batchB = (long long)H_*L_;
  pa.out_bf = hbuf; pa.K = L_;
  gemm_kernel<M_AV,false><<<dim3(H_/128, L_/128, B_), 256, 0, stream>>>(pa);

  // out = res + u*(silu(hx + (h*r)@h_w + h_b) - res)
  GemmP pf = {}; pf.A = hbuf; pf.A2 = rb; pf.Bt = hwt; pf.lda = H_; pf.ldb = H_;
  pf.bias = h_b; pf.hx = hxb; pf.u = ubuf; pf.xres = x; pf.out_f = out; pf.K = H_;
  gemm_kernel<M_FINAL,true><<<dim3(D_/128, (int)(LB/128), 1), 256, 0, stream>>>(pf);
}
